// DiverseGATConv_43430709297252
// MI455X (gfx1250) — matrix operand order repaired and checked by reading, not compiled, not run on hardware
//
#include <hip/hip_runtime.h>
#include <hip/hip_bf16.h>
#include <math.h>

#define N_NODES   100000
#define N_EDG     1600000
#define E_TOT     (N_EDG + N_NODES)   // 1,700,000 (self loops appended)
#define HEADS     4
#define OUTC      32
#define HC        128                 // HEADS*OUTC
#define NEG_SLOPE 0.2f
#define EPS_SOFT  1e-16f
#define EPS_COS   1e-8f

#define ROWS_PB   80                  // rows per block; 100000 = 1250 * 80
#define RT_PB     5                   // 16-row tiles per block

typedef __attribute__((ext_vector_type(16))) _Float16 v16h;
typedef __attribute__((ext_vector_type(8)))  _Float16 v8h;
typedef __attribute__((ext_vector_type(2)))  _Float16 v2h;
typedef __attribute__((ext_vector_type(8)))  float    v8f;

// -------- float atomics ------------------------------------------------------
__device__ __forceinline__ void atomicMaxF32(float* addr, float v) {
  // monotonic int-ordering trick; init value must be -inf
  if (v >= 0.0f) atomicMax((int*)addr, __float_as_int(v));
  else           atomicMin((unsigned int*)addr, __float_as_uint(v));
}
__device__ __forceinline__ void atomicAddF32(float* addr, float v) {
  unsafeAtomicAdd(addr, v);   // lowers to global_atomic_add_f32
}

// -------- Kernel 1: h = x @ W^T via WMMA, fused s_i/s_j ----------------------
// block = 256 threads (8 waves). Block owns 80 rows x 128 cols of h.
// W is staged once to LDS as f16; each wave keeps its B fragments resident in
// VGPRs across the 5 row tiles, so the inner loop is 2x ds_load_b128 + v_wmma.
__global__ __launch_bounds__(256) void k_gemm(
    const float* __restrict__ x, const float* __restrict__ W,
    const float* __restrict__ att,
    float* __restrict__ hbuf, float* __restrict__ si, float* __restrict__ sj) {
  __shared__ __align__(32) _Float16 lw[HC * HC];  // 32 KB: W as f16, [n][k]
  __shared__ __align__(32) float lxf[16 * HC];    // 8 KB: x tile (f16 view) / h tile (f32)
  __shared__ float latt[2 * HEADS * OUTC];        // 1 KB: att vectors
  _Float16* lxh = (_Float16*)lxf;

  const int t = threadIdx.x;
  const int rowBase = blockIdx.x * ROWS_PB;

  // stage W -> f16 LDS (coalesced; shared by all waves and all row tiles)
  #pragma unroll
  for (int i = 0; i < 32; ++i) {
    const int idx2 = t + 256 * i;                   // float2 index
    const float2 wv = ((const float2*)W)[idx2];
    ((v2h*)lw)[idx2] = (v2h){(_Float16)wv.x, (_Float16)wv.y};
  }
  latt[t] = att[t];                                 // exactly 256 floats
  __syncthreads();

  const int wave = t >> 5;
  const int lane = t & 31;
  const int hf   = lane >> 4;       // lane half select
  const int l15  = lane & 15;
  const int nCol = wave * 16 + l15; // output column 0..127

  // B 32x16 fragments (this code's contiguous k order, same as A): lane: N=l15, elem e -> K = kb + 16*hf + e
  v16h bfrag[4];
  #pragma unroll
  for (int kk = 0; kk < 4; ++kk)
    bfrag[kk] = *(const v16h*)&lw[nCol * HC + kk * 32 + hf * 16];

  #pragma unroll
  for (int rt = 0; rt < RT_PB; ++rt) {
    const int row0 = rowBase + rt * 16;
    __syncthreads();                                // lxf readers of prev tile done
    // stage 16x128 x tile as f16 (coalesced float2 loads, packed converts)
    {
      const float2* xs = (const float2*)(x + row0 * HC);
      #pragma unroll
      for (int i = 0; i < 4; ++i) {
        const int idx2 = t + 256 * i;
        const float2 v = xs[idx2];
        ((v2h*)lxh)[idx2] = (v2h){(_Float16)v.x, (_Float16)v.y};
      }
    }
    __syncthreads();

    v8f acc = {};
    #pragma unroll
    for (int kk = 0; kk < 4; ++kk) {                // K = 128 in steps of 32
      const int kb = kk * 32;
      // A 16x32 f16 layout: elem e -> K=kb+16*hf+e (matches B)
      const v16h a = *(const v16h*)&lxh[l15 * HC + kb + hf * 16];
      acc = __builtin_amdgcn_wmma_f32_16x16x32_f16(
          false, a, false, bfrag[kk], (short)0, acc, false, false);
    }
    __syncthreads();
    // D layout: lane l, vgpr v -> row hf*8+v, col nCol; park f32 tile in LDS
    #pragma unroll
    for (int v = 0; v < 8; ++v) lxf[(hf * 8 + v) * HC + nCol] = acc[v];
    __syncthreads();

    // coalesced store of h tile
    #pragma unroll
    for (int i = 0; i < 8; ++i) {
      const int idx = t + 256 * i;
      hbuf[row0 * HC + idx] = lxf[idx];
    }
    // fused per-node attention scalars: 64 threads, one (row, head) each
    if (t < 64) {
      const int r = t >> 2, hh = t & 3;
      float a_i = 0.f, a_j = 0.f;
      #pragma unroll
      for (int c = 0; c < OUTC; ++c) {
        const float hv = lxf[r * HC + hh * OUTC + c];
        a_i += hv * latt[hh * 2 * OUTC + c];
        a_j += hv * latt[hh * 2 * OUTC + OUTC + c];
      }
      si[(row0 + r) * HEADS + hh] = a_i;
      sj[(row0 + r) * HEADS + hh] = a_j;
    }
  }
}

// -------- Kernel 2: init accumulators (graph-replay safe) --------------------
__global__ void k_init(const float* __restrict__ bias, float* __restrict__ out,
                       float* __restrict__ emax, float* __restrict__ denom,
                       float* __restrict__ gram) {
  const int idx = blockIdx.x * blockDim.x + threadIdx.x;
  if (idx < N_NODES * HC) out[idx] = bias[idx & (HC - 1)];
  if (idx < N_NODES * HEADS) {
    emax[idx]  = -__builtin_inff();
    denom[idx] = 0.f;
  }
  if (idx < 16) gram[idx] = 0.f;
}

// -------- edge decode helper -------------------------------------------------
__device__ __forceinline__ void edge_rc(const int* __restrict__ ei, int e,
                                        int& r, int& c) {
  if (e < N_EDG) { r = ei[e]; c = ei[N_EDG + e]; }
  else           { r = c = e - N_EDG; }          // appended self loop
}

// -------- Kernel 3: leaky-relu logits + segment max --------------------------
__global__ void k_logits_max(const int* __restrict__ ei,
                             const float* __restrict__ si,
                             const float* __restrict__ sj,
                             float* __restrict__ emax) {
  const int e = blockIdx.x * blockDim.x + threadIdx.x;
  if (e >= E_TOT) return;
  int r, c; edge_rc(ei, e, r, c);
  #pragma unroll
  for (int hh = 0; hh < HEADS; ++hh) {
    float v = si[r * HEADS + hh] + sj[c * HEADS + hh];
    v = (v >= 0.f) ? v : NEG_SLOPE * v;
    atomicMaxF32(&emax[r * HEADS + hh], v);
  }
}

// -------- Kernel 4: exp + segment sum ----------------------------------------
__global__ void k_exp_sum(const int* __restrict__ ei,
                          const float* __restrict__ si,
                          const float* __restrict__ sj,
                          const float* __restrict__ emax,
                          float* __restrict__ eexp, float* __restrict__ denom) {
  const int e = blockIdx.x * blockDim.x + threadIdx.x;
  if (e >= E_TOT) return;
  int r, c; edge_rc(ei, e, r, c);
  #pragma unroll
  for (int hh = 0; hh < HEADS; ++hh) {
    float v = si[r * HEADS + hh] + sj[c * HEADS + hh];
    v = (v >= 0.f) ? v : NEG_SLOPE * v;
    const float ex = __expf(v - emax[r * HEADS + hh]);
    eexp[e * HEADS + hh] = ex;
    atomicAddF32(&denom[r * HEADS + hh], ex);
  }
}

// -------- Kernel 5: message scatter (roofline-dominant) ----------------------
// one wave per edge: lane owns 4 channels -> float4 gather + 4 fp32 atomics
__global__ __launch_bounds__(256) void k_scatter(
    const int* __restrict__ ei, const float* __restrict__ hbuf,
    const float* __restrict__ eexp, const float* __restrict__ denom,
    float* __restrict__ out) {
  const int gw   = (blockIdx.x * blockDim.x + threadIdx.x) >> 5;
  const int lane = threadIdx.x & 31;
  if (gw >= E_TOT) return;
  int r, c; edge_rc(ei, gw, r, c);
  const int hh = lane >> 3;                         // head for these 4 channels
  const float alpha =
      eexp[gw * HEADS + hh] / (denom[r * HEADS + hh] + EPS_SOFT);
  const float4 hv = *(const float4*)&hbuf[c * HC + lane * 4];
  float* op = &out[r * HC + lane * 4];
  atomicAddF32(op + 0, hv.x * alpha);
  atomicAddF32(op + 1, hv.y * alpha);
  atomicAddF32(op + 2, hv.z * alpha);
  atomicAddF32(op + 3, hv.w * alpha);
}

// -------- Kernel 6: gram = alpha^T alpha (block partials -> 16 atomics) ------
__global__ __launch_bounds__(256) void k_gram(
    const int* __restrict__ ei, const float* __restrict__ eexp,
    const float* __restrict__ denom, float* __restrict__ gram) {
  __shared__ float sg[16];
  if (threadIdx.x < 16) sg[threadIdx.x] = 0.f;
  __syncthreads();
  float g[16];
  #pragma unroll
  for (int i = 0; i < 16; ++i) g[i] = 0.f;
  for (int e = blockIdx.x * blockDim.x + threadIdx.x; e < E_TOT;
       e += gridDim.x * blockDim.x) {
    int r, c; edge_rc(ei, e, r, c); (void)c;
    float a[HEADS];
    #pragma unroll
    for (int hh = 0; hh < HEADS; ++hh)
      a[hh] = eexp[e * HEADS + hh] / (denom[r * HEADS + hh] + EPS_SOFT);
    #pragma unroll
    for (int i = 0; i < HEADS; ++i)
      #pragma unroll
      for (int j = 0; j < HEADS; ++j) g[i * 4 + j] += a[i] * a[j];
  }
  #pragma unroll
  for (int i = 0; i < 16; ++i) atomicAdd(&sg[i], g[i]);   // ds_add_f32
  __syncthreads();
  if (threadIdx.x < 16) atomicAddF32(&gram[threadIdx.x], sg[threadIdx.x]);
}

// -------- Kernel 7: diversity loss epilogue ----------------------------------
__global__ void k_loss(const float* __restrict__ gram, float* __restrict__ lo) {
  if (threadIdx.x == 0) {
    float nrm[HEADS];
    #pragma unroll
    for (int hh = 0; hh < HEADS; ++hh) nrm[hh] = sqrtf(gram[hh * 4 + hh]);
    float s = 0.f;
    #pragma unroll
    for (int i = 0; i < HEADS; ++i)
      #pragma unroll
      for (int j = 0; j < HEADS; ++j)
        if (i != j) s += gram[i * 4 + j] / fmaxf(nrm[i] * nrm[j], EPS_COS);
    lo[0] = (s / 16.0f) * 0.1f;    // mean over HxH (diag zeroed) * DIV_WEIGHT
  }
}

// -------- launcher -----------------------------------------------------------
extern "C" void kernel_launch(void* const* d_in, const int* in_sizes, int n_in,
                              void* d_out, int out_size, void* d_ws,
                              size_t ws_size, hipStream_t stream) {
  const float* x    = (const float*)d_in[0];
  const int*   ei   = (const int*)d_in[1];
  const float* W    = (const float*)d_in[2];
  const float* att  = (const float*)d_in[3];
  const float* bias = (const float*)d_in[4];

  float* out  = (float*)d_out;
  float* loss = out + (size_t)N_NODES * HC;

  float* ws    = (float*)d_ws;
  float* hbuf  = ws;                                  // 12,800,000 f32
  float* si    = hbuf  + (size_t)N_NODES * HC;        //    400,000
  float* sj    = si    + (size_t)N_NODES * HEADS;     //    400,000
  float* emax  = sj    + (size_t)N_NODES * HEADS;     //    400,000
  float* denom = emax  + (size_t)N_NODES * HEADS;     //    400,000
  float* eexp  = denom + (size_t)N_NODES * HEADS;     //  6,800,000
  float* gram  = eexp  + (size_t)E_TOT * HEADS;       //         16

  k_gemm<<<N_NODES / ROWS_PB, 256, 0, stream>>>(x, W, att, hbuf, si, sj);
  k_init<<<(N_NODES * HC + 255) / 256, 256, 0, stream>>>(bias, out, emax,
                                                         denom, gram);
  k_logits_max<<<(E_TOT + 255) / 256, 256, 0, stream>>>(ei, si, sj, emax);
  k_exp_sum<<<(E_TOT + 255) / 256, 256, 0, stream>>>(ei, si, sj, emax, eexp,
                                                     denom);
  k_scatter<<<(E_TOT * 32 + 255) / 256, 256, 0, stream>>>(ei, hbuf, eexp,
                                                          denom, out);
  k_gram<<<2048, 256, 0, stream>>>(ei, eexp, denom, gram);
  k_loss<<<1, 32, 0, stream>>>(gram, loss);
}
